// MultiHeadAttention_80745385165007
// MI455X (gfx1250) — matrix operand order repaired and checked by reading, not compiled, not run on hardware
//
#include <hip/hip_runtime.h>
#include <math.h>
#include <stdint.h>

// ---------------------------------------------------------------------------
// MHA for MI455X (gfx1250, wave32).  All matmuls via v_wmma_f32_16x16x32_bf16.
// Flash-style attention: scores never touch HBM (saves ~1 GB of traffic vs the
// naive 4x16x2048x2048 f32 score tensor; ideal traffic is ~135 MB @ 23.3 TB/s).
// K tiles staged global->LDS through the gfx1250 async-copy path (ASYNCcnt).
// ---------------------------------------------------------------------------

#define B_N 4
#define S_N 2048
#define H_N 16
#define D_H 64
#define D_M 1024
#define QTILES (S_N / 16) /* 128 */

typedef __attribute__((ext_vector_type(16))) __bf16 v16bf;
typedef __attribute__((ext_vector_type(8)))  __bf16 v8bf;
typedef __attribute__((ext_vector_type(8)))  float  v8f;

#if __has_builtin(__builtin_amdgcn_global_load_async_to_lds_b128)
#define HAVE_ASYNC_LDS 1
typedef int v4i_gl __attribute__((ext_vector_type(4)));
typedef __attribute__((address_space(1))) v4i_gl gas_v4i;
typedef __attribute__((address_space(3))) v4i_gl las_v4i;
#else
#define HAVE_ASYNC_LDS 0
#endif

static __device__ __forceinline__ void wait_async0() {
#if __has_builtin(__builtin_amdgcn_s_wait_asynccnt)
  __builtin_amdgcn_s_wait_asynccnt(0);
  asm volatile("" ::: "memory");
#else
  asm volatile("s_wait_asynccnt 0x0" ::: "memory");
#endif
}

static __device__ __forceinline__ v8f wmma_bf16(v16bf a, v16bf b, v8f c) {
  // (neg_a, A, neg_b, B, c_mod, C, reuse_a, reuse_b)
  return __builtin_amdgcn_wmma_f32_16x16x32_bf16(false, a, false, b, (short)0, c,
                                                 false, false);
}

// A-matrix tile (16 rows x 32 K) from a row-major source, ld in elements.
// ISA layout: lane m (0-15) holds row m, K = 0..7 (v0-3) & 16..23 (v4-7);
// lane m+16 holds row m, K = 8..15 & 24..31.
static __device__ __forceinline__ v16bf load_a16x32_bf16(const __bf16* p, int ld) {
  const int lane = threadIdx.x & 31;
  const int m = lane & 15;
  const int ko = (lane >> 4) << 3; // 0 or 8
  const __bf16* row = p + m * ld;
  v8bf lo = *(const v8bf*)(row + ko);
  v8bf hi = *(const v8bf*)(row + 16 + ko);
  v16bf r;
#pragma unroll
  for (int i = 0; i < 8; ++i) { r[i] = lo[i]; r[8 + i] = hi[i]; }
  return r;
}

// Same A layout but converting f32 source -> bf16 on the fly.
static __device__ __forceinline__ v16bf load_a16x32_f32(const float* p, int ld) {
  const int lane = threadIdx.x & 31;
  const int m = lane & 15;
  const int ko = (lane >> 4) << 3;
  const float* row = p + m * ld;
  v16bf r;
#pragma unroll
  for (int i = 0; i < 8; ++i) {
    r[i]     = (__bf16)row[ko + i];
    r[8 + i] = (__bf16)row[16 + ko + i];
  }
  return r;
}

// B-matrix tile (32 K x 16 N) where COLUMN n of B is ROW n of `p` (i.e. we are
// computing A x P^T with P row-major).  ISA layout: lane n (0-15) holds col n,
// K = 0..7 & 16..23; lane n+16 holds col n, K = 8..15 & 24..31.
static __device__ __forceinline__ v16bf load_b32x16_rows(const __bf16* p, int ld) {
  const int lane = threadIdx.x & 31;
  const int n = lane & 15;
  const int kh = (lane >> 4) << 3; // 0 or 8
  const __bf16* row = p + n * ld + kh;
  v8bf lo = *(const v8bf*)(row);
  v8bf hi = *(const v8bf*)(row + 16);
  v16bf r;
#pragma unroll
  for (int i = 0; i < 8; ++i) { r[i] = lo[i]; r[8 + i] = hi[i]; }
  return r;
}

// Reductions across the 16 lanes of a half-wave (rows of a C/D tile span
// lanes 0-15 / 16-31; xor masks 1,2,4,8 stay inside each half).
static __device__ __forceinline__ float redmax16(float v) {
#pragma unroll
  for (int m = 1; m <= 8; m <<= 1) v = fmaxf(v, __shfl_xor(v, m, 32));
  return v;
}
static __device__ __forceinline__ float redsum16(float v) {
#pragma unroll
  for (int m = 1; m <= 8; m <<= 1) v += __shfl_xor(v, m, 32);
  return v;
}

// ---------------------------------------------------------------------------
// Kernel 1: convert weights f32 -> bf16 (wq,wk,wv: 64x64 ; wo: 1024x1024)
// ---------------------------------------------------------------------------
__global__ void cvt_weights_kernel(const float* __restrict__ wq,
                                   const float* __restrict__ wk,
                                   const float* __restrict__ wv,
                                   const float* __restrict__ wo,
                                   __bf16* __restrict__ wqb, __bf16* __restrict__ wkb,
                                   __bf16* __restrict__ wvb, __bf16* __restrict__ wob) {
  int i = blockIdx.x * blockDim.x + threadIdx.x;
  if (i < 4096)       { wqb[i] = (__bf16)wq[i]; return; }
  if (i < 8192)       { wkb[i - 4096] = (__bf16)wk[i - 4096]; return; }
  if (i < 12288)      { wvb[i - 8192] = (__bf16)wv[i - 8192]; return; }
  int j = i - 12288;
  if (j < D_M * D_M)  { wob[j] = (__bf16)wo[j]; }
}

// ---------------------------------------------------------------------------
// Kernel 2: per-head QKV projections.  One wave per (which, b, h, 16-row tile).
// q/k stored [b][h][s][64] bf16 ; v stored TRANSPOSED [b][h][64][s] bf16 so the
// attention PV B-operand is a contiguous row load.
// ---------------------------------------------------------------------------
__global__ void qkv_proj_kernel(const float* __restrict__ values,
                                const float* __restrict__ keys,
                                const float* __restrict__ query,
                                const __bf16* __restrict__ wqb,
                                const __bf16* __restrict__ wkb,
                                const __bf16* __restrict__ wvb,
                                const float* __restrict__ bq,
                                const float* __restrict__ bk,
                                const float* __restrict__ bv,
                                __bf16* __restrict__ q_ws,
                                __bf16* __restrict__ k_ws,
                                __bf16* __restrict__ vT_ws) {
  const int wave = (blockIdx.x * blockDim.x + threadIdx.x) >> 5;
  const int lane = threadIdx.x & 31;
  const int tiles = B_N * H_N * QTILES;
  const int which = wave / tiles; // 0=q 1=k 2=v (wave-uniform)
  int rem = wave - which * tiles;
  const int b = rem / (H_N * QTILES); rem -= b * (H_N * QTILES);
  const int h = rem / QTILES;
  const int s0 = (rem - h * QTILES) * 16;

  const float*  xin  = (which == 0) ? query : (which == 1) ? keys : values;
  const __bf16* w    = (which == 0) ? wqb   : (which == 1) ? wkb  : wvb;
  const float*  bias = (which == 0) ? bq    : (which == 1) ? bk   : bv;

  const float* xrow = xin + ((size_t)b * S_N + s0) * D_M + h * D_H;
  v16bf a0 = load_a16x32_f32(xrow, D_M);       // K = 0..31 of head_dim
  v16bf a1 = load_a16x32_f32(xrow + 32, D_M);  // K = 32..63

  const int n  = lane & 15;
  const int mb = (lane >> 4) << 3;
#pragma unroll
  for (int t = 0; t < 4; ++t) {
    const int n0 = t * 16;
    v16bf b0 = load_b32x16_rows(w + n0 * D_H, D_H);
    v16bf b1 = load_b32x16_rows(w + n0 * D_H + 32, D_H);
    v8f c = {};
    c = wmma_bf16(a0, b0, c);
    c = wmma_bf16(a1, b1, c);
    const float bs = bias[n0 + n];
    if (which < 2) {
      __bf16* dst = ((which == 0) ? q_ws : k_ws) +
                    ((size_t)b * H_N + h) * (size_t)S_N * D_H;
#pragma unroll
      for (int j = 0; j < 8; ++j)
        dst[(size_t)(s0 + mb + j) * D_H + n0 + n] = (__bf16)(c[j] + bs);
    } else {
      __bf16* dst = vT_ws + ((size_t)b * H_N + h) * (size_t)D_H * S_N;
#pragma unroll
      for (int j = 0; j < 8; ++j)
        dst[(size_t)(n0 + n) * S_N + (s0 + mb + j)] = (__bf16)(c[j] + bs);
    }
  }
}

// ---------------------------------------------------------------------------
// Kernel 3: flash attention.  One wave per (b, h, 16-query tile); loop over
// 32-key blocks with online softmax.  K block (32x64 bf16 = 4 KB) is staged
// into LDS via gfx1250 async global->LDS copies (ASYNCcnt); P is restaged
// through LDS (C-layout -> A-layout) for the PV matmul.
// ---------------------------------------------------------------------------
__global__ void flash_attn_kernel(const __bf16* __restrict__ q_ws,
                                  const __bf16* __restrict__ k_ws,
                                  const __bf16* __restrict__ vT_ws,
                                  __bf16* __restrict__ o_ws) {
  __shared__ alignas(16) __bf16 pstage[4][16 * 32]; // per-wave P staging (1 KB)
#if HAVE_ASYNC_LDS
  __shared__ alignas(16) __bf16 kstage[4][32 * D_H]; // per-wave K staging (4 KB)
#endif
  const int wslot = threadIdx.x >> 5;
  const int wave = (blockIdx.x * blockDim.x + threadIdx.x) >> 5;
  const int lane = threadIdx.x & 31;
  const int b = wave / (H_N * QTILES);
  int rem = wave - b * (H_N * QTILES);
  const int h = rem / QTILES;
  const int q0 = (rem - h * QTILES) * 16;

  const size_t head = ((size_t)b * H_N + h) * (size_t)S_N * D_H;
  const __bf16* qh = q_ws + head;
  const __bf16* kp = k_ws + head;
  const __bf16* vh = vT_ws + head; // [64][S_N] per head

  v16bf aq0 = load_a16x32_bf16(qh + (size_t)q0 * D_H, D_H);
  v16bf aq1 = load_a16x32_bf16(qh + (size_t)q0 * D_H + 32, D_H);

  const float scale = 0.03125f; // 1/sqrt(d_model=1024)
  float mrow[8], lrow[8], alpha[8];
  v8f acc[4];
  {
    v8f z = {};
#pragma unroll
    for (int t = 0; t < 4; ++t) acc[t] = z;
  }
#pragma unroll
  for (int j = 0; j < 8; ++j) { mrow[j] = -1e30f; lrow[j] = 0.f; }

  const int n  = lane & 15;
  const int mb = (lane >> 4) << 3;

#if HAVE_ASYNC_LDS
  // Per-lane base addresses for the async K copy: lane l owns row l (128 B).
  const uint32_t klds_base =
      (uint32_t)(uintptr_t)(const void*)&kstage[wslot][0] + (uint32_t)lane * 128u;
#endif

  for (int k0 = 0; k0 < S_N; k0 += 32) {
    if (k0 + 32 < S_N) { // wave-uniform guard; emits global_prefetch_b8
      __builtin_prefetch(kp + (size_t)(k0 + 32) * D_H, 0, 0);
      __builtin_prefetch(vh + (k0 + 32), 0, 0);
    }

#if HAVE_ASYNC_LDS
    // Stage the 32x64 bf16 K block into LDS: 8 async b128 ops x 32 lanes.
    {
      const char* gk = (const char*)(kp + (size_t)k0 * D_H) + (size_t)lane * 128u;
      asm volatile("" ::: "memory");
#pragma unroll
      for (int i = 0; i < 8; ++i) {
        __builtin_amdgcn_global_load_async_to_lds_b128(
            (gas_v4i*)(uintptr_t)(gk + i * 16),
            (las_v4i*)(uintptr_t)(klds_base + (uint32_t)(i * 16)), 0, 0);
      }
      wait_async0();
    }
    const __bf16* kblk = &kstage[wslot][0];
#else
    const __bf16* kblk = kp + (size_t)k0 * D_H;
#endif

    // S = Q K^T for two 16-key column tiles (K-dim = head_dim = 64 -> 2 steps)
    v8f s0 = {}, s1 = {};
    s0 = wmma_bf16(aq0, load_b32x16_rows(kblk, D_H), s0);
    s0 = wmma_bf16(aq1, load_b32x16_rows(kblk + 32, D_H), s0);
    s1 = wmma_bf16(aq0, load_b32x16_rows(kblk + 16 * D_H, D_H), s1);
    s1 = wmma_bf16(aq1, load_b32x16_rows(kblk + 16 * D_H + 32, D_H), s1);

    // Online softmax update; C/D row m=mb+j lives in VGPR j across 16 lanes.
#pragma unroll
    for (int j = 0; j < 8; ++j) {
      float e0 = s0[j] * scale;
      float e1 = s1[j] * scale;
      float mnew = fmaxf(mrow[j], redmax16(fmaxf(e0, e1)));
      float a_ = __expf(mrow[j] - mnew);
      e0 = __expf(e0 - mnew);
      e1 = __expf(e1 - mnew);
      lrow[j] = lrow[j] * a_ + redsum16(e0 + e1);
      mrow[j] = mnew;
      alpha[j] = a_;
      pstage[wslot][(mb + j) * 32 + n]      = (__bf16)e0;
      pstage[wslot][(mb + j) * 32 + 16 + n] = (__bf16)e1;
    }
    asm volatile("s_wait_dscnt 0x0" ::: "memory"); // P fully in LDS (same wave)
    v16bf pa = load_a16x32_bf16(&pstage[wslot][0], 32);

#pragma unroll
    for (int t = 0; t < 4; ++t)
#pragma unroll
      for (int j = 0; j < 8; ++j) acc[t][j] *= alpha[j];

    // O += P V ; B operand = rows of V^T (contiguous, stride S_N)
#pragma unroll
    for (int t = 0; t < 4; ++t) {
      v16bf bv_ = load_b32x16_rows(vh + (size_t)(t * 16) * S_N + k0, S_N);
      acc[t] = wmma_bf16(pa, bv_, acc[t]);
    }
  }

  // Normalize and write O as bf16 into [b][s][h*64+d] for the final GEMM.
  __bf16* orow = o_ws + ((size_t)b * S_N + q0) * D_M + h * D_H;
#pragma unroll
  for (int t = 0; t < 4; ++t)
#pragma unroll
    for (int j = 0; j < 8; ++j)
      orow[(size_t)(mb + j) * D_M + t * 16 + n] = (__bf16)(acc[t][j] / lrow[j]);
}

// ---------------------------------------------------------------------------
// Kernel 4: output projection  out = O @ wo^T + bo  (8192 x 1024 x 1024).
// One wave per 16x64 output tile; K-loop of 32-wide WMMA steps.
// ---------------------------------------------------------------------------
__global__ void out_proj_kernel(const __bf16* __restrict__ o_ws,
                                const __bf16* __restrict__ wob,
                                const float* __restrict__ bo,
                                float* __restrict__ out) {
  const int wave = (blockIdx.x * blockDim.x + threadIdx.x) >> 5;
  const int lane = threadIdx.x & 31;
  const int m0 = (wave >> 4) * 16;
  const int n0g = (wave & 15) * 64;
  const __bf16* arow = o_ws + (size_t)m0 * D_M;

  v8f acc[4];
  {
    v8f z = {};
#pragma unroll
    for (int t = 0; t < 4; ++t) acc[t] = z;
  }
  for (int k0 = 0; k0 < D_M; k0 += 32) {
    v16bf a = load_a16x32_bf16(arow + k0, D_M);
#pragma unroll
    for (int t = 0; t < 4; ++t) {
      v16bf bb = load_b32x16_rows(wob + (size_t)(n0g + t * 16) * D_M + k0, D_M);
      acc[t] = wmma_bf16(a, bb, acc[t]);
    }
  }
  const int n = lane & 15;
  const int mb = (lane >> 4) << 3;
#pragma unroll
  for (int t = 0; t < 4; ++t) {
    const float bs = bo[n0g + t * 16 + n];
#pragma unroll
    for (int j = 0; j < 8; ++j)
      out[(size_t)(m0 + mb + j) * D_M + n0g + t * 16 + n] = acc[t][j] + bs;
  }
}

// ---------------------------------------------------------------------------
extern "C" void kernel_launch(void* const* d_in, const int* in_sizes, int n_in,
                              void* d_out, int out_size, void* d_ws, size_t ws_size,
                              hipStream_t stream) {
  (void)in_sizes; (void)n_in; (void)out_size; (void)ws_size;
  // setup_inputs order: values, keys, query, wq, bq, wk, bk, wv, bv, wo, bo
  const float* values = (const float*)d_in[0];
  const float* keys   = (const float*)d_in[1];
  const float* query  = (const float*)d_in[2];
  const float* wq = (const float*)d_in[3];
  const float* bq = (const float*)d_in[4];
  const float* wk = (const float*)d_in[5];
  const float* bk = (const float*)d_in[6];
  const float* wv = (const float*)d_in[7];
  const float* bv = (const float*)d_in[8];
  const float* wo = (const float*)d_in[9];
  const float* bo = (const float*)d_in[10];
  float* out = (float*)d_out;

  // Workspace carve-up (~66 MB total, 256B-aligned slices)
  char* ws = (char*)d_ws;
  __bf16* wqb = (__bf16*)(ws);
  __bf16* wkb = (__bf16*)(ws + 8192);
  __bf16* wvb = (__bf16*)(ws + 16384);
  __bf16* wob = (__bf16*)(ws + 24576);
  size_t off = 24576 + (size_t)D_M * D_M * 2;                    // 2121728
  const size_t per = (size_t)B_N * H_N * S_N * D_H * 2;          // 16 MB each
  __bf16* q_ws  = (__bf16*)(ws + off); off += per;
  __bf16* k_ws  = (__bf16*)(ws + off); off += per;
  __bf16* vT_ws = (__bf16*)(ws + off); off += per;
  __bf16* o_ws  = (__bf16*)(ws + off);

  // 1) weights -> bf16   (12288 + 1048576 = 1060864 threads)
  cvt_weights_kernel<<<4144, 256, 0, stream>>>(wq, wk, wv, wo, wqb, wkb, wvb, wob);
  // 2) QKV projections: 3*4*16*128 = 24576 waves
  qkv_proj_kernel<<<6144, 128, 0, stream>>>(values, keys, query, wqb, wkb, wvb,
                                            bq, bk, bv, q_ws, k_ws, vT_ws);
  // 3) flash attention: 4*16*128 = 8192 waves
  flash_attn_kernel<<<2048, 128, 0, stream>>>(q_ws, k_ws, vT_ws, o_ws);
  // 4) output projection: 512*16 = 8192 waves
  out_proj_kernel<<<2048, 128, 0, stream>>>(o_ws, wob, bo, out);
}
